// SlidingWindowSelfAttention_67104569033320
// MI455X (gfx1250) — matrix operand order repaired and checked by reading, not compiled, not run on hardware
//
#include <hip/hip_runtime.h>

// Sliding-window GQA self-attention for MI455X (gfx1250, wave32, WMMA).
// B=2, S=2048, D=1024, HQ=16, HKV=4, HD=64, WINDOW=256 (|q-k| <= 128).
//
// Pipeline (bf16 storage, f32 accum via v_wmma_f32_16x16x32_bf16):
//   1) convert x -> bf16; pack [Wq|Wk|Wv] transposed -> bf16; Wo transposed -> bf16
//   2) fused QKV GEMM   (4096x1536x1024), wave tile 16x128
//   3) RoPE on Q and K heads; Q pre-scaled by 1/sqrt(HD) = 0.125 (exact)
//   4) flash-style windowed attention; V staged via global_load_async_to_lds_b128;
//      softmax reductions via DPP16 butterflies (no ds_bpermute)
//   5) output GEMM      (4096x1024x1024) -> f32 out

#define BD   2
#define SD   2048
#define DD   1024
#define HQD  16
#define HKVD 4
#define HDD  64
#define ROWS (BD * SD)          // 4096
#define NQKV 1536               // HQ*HD + 2*HKV*HD
#define WHALF 128

typedef __attribute__((ext_vector_type(16))) __bf16 v16bf;
typedef __attribute__((ext_vector_type(8)))  float  v8f;

union Frag {
    v16bf v;
    uint4 q[2];
    unsigned short us[16];
};

__device__ __forceinline__ float bf2f(unsigned short u) {
    unsigned int x = ((unsigned int)u) << 16;
    float f;
    __builtin_memcpy(&f, &x, 4);
    return f;
}
__device__ __forceinline__ unsigned short f2bf(float f) {
    unsigned int x;
    __builtin_memcpy(&x, &f, 4);
    unsigned int r = (x + 0x7FFFu + ((x >> 16) & 1u)) >> 16;
    return (unsigned short)r;
}

// DPP16 lane-permute move (VALU only, no LDS traffic).
template <int CTRL>
__device__ __forceinline__ float dppf(float x) {
    int i = __float_as_int(x);
    i = __builtin_amdgcn_update_dpp(i, i, CTRL, 0xF, 0xF, true);
    return __int_as_float(i);
}
// 16-lane butterfly reductions: ^1, ^2 via quad_perm, then half-row mirror, row mirror.
// (After each step every sub-group is uniform, so reversal partners complete the tree.)
__device__ __forceinline__ float rowmax16(float v) {
    v = fmaxf(v, dppf<0xB1>(v));    // quad_perm [1,0,3,2]
    v = fmaxf(v, dppf<0x4E>(v));    // quad_perm [2,3,0,1]
    v = fmaxf(v, dppf<0x141>(v));   // row_half_mirror
    v = fmaxf(v, dppf<0x140>(v));   // row_mirror
    return v;
}
__device__ __forceinline__ float rowsum16(float v) {
    v += dppf<0xB1>(v);
    v += dppf<0x4E>(v);
    v += dppf<0x141>(v);
    v += dppf<0x140>(v);
    return v;
}

// ---------------- conversion / packing kernels ----------------

__global__ void k_f32_to_bf16(const float* __restrict__ in, unsigned short* __restrict__ out, int n) {
    int i = blockIdx.x * blockDim.x + threadIdx.x;
    if (i < n) out[i] = f2bf(in[i]);
}

// WqkvT[n][k] (n < 1536, k < 1024): transposed + fused weight, row-major over n.
__global__ void k_pack_wqkv(const float* __restrict__ Wq, const float* __restrict__ Wk,
                            const float* __restrict__ Wv, unsigned short* __restrict__ WT) {
    int i = blockIdx.x * blockDim.x + threadIdx.x;
    if (i >= NQKV * DD) return;
    int n = i / DD, k = i % DD;
    float v;
    if (n < 1024)      v = Wq[(size_t)k * 1024 + n];
    else if (n < 1280) v = Wk[(size_t)k * 256 + (n - 1024)];
    else               v = Wv[(size_t)k * 256 + (n - 1280)];
    WT[i] = f2bf(v);
}

__global__ void k_pack_wo(const float* __restrict__ Wo, unsigned short* __restrict__ WT) {
    int i = blockIdx.x * blockDim.x + threadIdx.x;
    if (i >= DD * DD) return;
    int n = i >> 10, k = i & 1023;
    WT[i] = f2bf(Wo[(size_t)k * 1024 + n]);
}

// ---------------- RoPE (in place on fused QKV, cols 0..1279 = 20 heads) ----------------
// Q heads (0..15) are additionally scaled by 1/sqrt(HD) = 0.125 (exact power of two),
// which removes the score scaling from the attention inner loop.

__global__ void k_rope(unsigned short* __restrict__ qkv) {
    int i = blockIdx.x * blockDim.x + threadIdx.x;
    const int TOT = ROWS * 20 * 32;   // rows * heads(Q16+K4) * pairs
    if (i >= TOT) return;
    int pair = i & 31;
    int head = (i >> 5) % 20;
    int row  = i / (32 * 20);
    int pos  = row & (SD - 1);
    unsigned short* p = qkv + (size_t)row * NQKV + head * HDD + pair * 2;
    float x1 = bf2f(p[0]), x2 = bf2f(p[1]);
    float inv = __expf(-((float)(2 * pair) / 64.f) * 9.210340371976184f);  // base^(-2i/hd)
    float ang = (float)pos * inv;
    float sn, cs;
    __sincosf(ang, &sn, &cs);
    float s8 = (head < HQD) ? 0.125f : 1.f;
    p[0] = f2bf((x1 * cs - x2 * sn) * s8);
    p[1] = f2bf((x1 * sn + x2 * cs) * s8);
}

// ---------------- generic bf16 WMMA GEMM ----------------
// C[M,N] = A[M,K] * B[K,N];  A row-major bf16 (ldA), BT = B^T row-major bf16 (ldB = K),
// wave computes a 16(M) x 128(N) tile; K-step 32 -> 8x v_wmma_f32_16x16x32_bf16 per A load.

template <int OUT_BF16>
__global__ void __launch_bounds__(256)
k_gemm(const unsigned short* __restrict__ A, const unsigned short* __restrict__ BT,
       void* __restrict__ Cv, int M, int N, int K, int ldA, int ldB, int ldC) {
    const int lane = threadIdx.x & 31;
    const int gw   = (blockIdx.x * blockDim.x + threadIdx.x) >> 5;
    const int tilesN = N >> 7;
    const int tm = gw / tilesN, tn = gw % tilesN;
    if (tm * 16 >= M) return;
    const int m  = lane & 15;
    const int hf = lane >> 4;

    v8f acc[8];
#pragma unroll
    for (int j = 0; j < 8; ++j) acc[j] = v8f{0, 0, 0, 0, 0, 0, 0, 0};

    const unsigned short* arow  = A  + (size_t)(tm * 16 + m) * ldA;
    const unsigned short* brow0 = BT + (size_t)(tn * 128 + m) * ldB;

    for (int k0 = 0; k0 < K; k0 += 32) {
        Frag a;
        // A-layout: element e -> K = (e>>3)*16 + (e&7) + hf*8  (two 16B runs)
        a.q[0] = *(const uint4*)(arow + k0 + hf * 8);
        a.q[1] = *(const uint4*)(arow + k0 + 16 + hf * 8);
#pragma unroll
        for (int j = 0; j < 8; ++j) {
            // B-layout: lane n = (lane&15), element e -> K = (e>>3)*16 + (e&7) + hf*8  (two 16B runs)
            const unsigned short* br = brow0 + (size_t)j * 16 * ldB + k0 + hf * 8;
            Frag bb;
            bb.q[0] = *(const uint4*)(br);
            bb.q[1] = *(const uint4*)(br + 16);
            acc[j] = __builtin_amdgcn_wmma_f32_16x16x32_bf16(
                false, a.v, false, bb.v, (short)0, acc[j], false, false);
        }
    }
    // C/D layout: vgpr r -> row = tm*16 + r + hf*8, col = tn*128 + j*16 + (lane&15)
#pragma unroll
    for (int j = 0; j < 8; ++j)
#pragma unroll
        for (int r = 0; r < 8; ++r) {
            size_t row = (size_t)(tm * 16 + r + hf * 8);
            size_t col = (size_t)(tn * 128 + j * 16 + m);
            float v = acc[j][r];
            if (OUT_BF16) ((unsigned short*)Cv)[row * ldC + col] = f2bf(v);
            else          ((float*)Cv)[row * ldC + col] = v;
        }
}

// ---------------- windowed flash attention ----------------
// one wave per (b, h, 16-query tile); 8 waves / 256-thread block.
// Per 32-key iteration:
//   - issue 8x global_load_async_to_lds_b128: V tile (32 keys x 64 hd bf16, 4KB) -> LDS
//   - K loads (coalesced b128) + 4 score WMMAs overlap the async copy
//   - online softmax; row reductions via DPP16 butterflies (VALU only)
//   - P transposed C-layout -> A-layout through per-wave LDS (s_wait_dscnt)
//   - s_wait_asynccnt 0, then 4 P*V WMMAs with V fragments gathered from LDS
// OOB note: last iteration may touch up to 15 rows past `qkv`; that memory is the
// adjacent `ctx` region of the workspace (finite bf16) and those keys are masked.

__global__ void __launch_bounds__(256)
k_attn(const unsigned short* __restrict__ qkv, unsigned short* __restrict__ ctx) {
    __shared__ unsigned short ldsP[8][16][32];   // per-wave 16(q) x 32(key) P tile
    __shared__ unsigned short ldsV[8][32][64];   // per-wave 32(key) x 64(hd) V tile

    const int lane = threadIdx.x & 31;
    const int wv   = threadIdx.x >> 5;
    const int gw   = blockIdx.x * 8 + wv;
    const int qt   = gw & 127;           // S/16 = 128 query tiles
    const int h    = (gw >> 7) & 15;
    const int b    = gw >> 11;
    const int kvh  = h >> 2;             // GQA: 4 Q heads per KV head
    const int m    = lane & 15;
    const int hf   = lane >> 4;
    const int q0   = qt * 16;

    // Q fragments (A-layout), hd chunks [0,32) and [32,64); Q pre-scaled by 0.125
    Frag qa0, qa1;
    {
        const unsigned short* qrow = qkv + (size_t)(b * SD + q0 + m) * NQKV + h * HDD;
        qa0.q[0] = *(const uint4*)(qrow + hf * 8);
        qa0.q[1] = *(const uint4*)(qrow + 16 + hf * 8);
        qa1.q[0] = *(const uint4*)(qrow + 32 + hf * 8);
        qa1.q[1] = *(const uint4*)(qrow + 48 + hf * 8);
    }

    float mi[8], li[8];
    v8f oacc[4];
#pragma unroll
    for (int r = 0; r < 8; ++r) { mi[r] = -1e30f; li[r] = 0.f; }
#pragma unroll
    for (int j = 0; j < 4; ++j) oacc[j] = v8f{0, 0, 0, 0, 0, 0, 0, 0};

    int kb_lo = q0 - WHALF; if (kb_lo < 0) kb_lo = 0;      // tile-aligned
    int kb_hi = q0 + 16 + WHALF; if (kb_hi > SD) kb_hi = SD;

    // strength-reduced pointers (advance by 32 key rows per iteration)
    const unsigned short* krow0 =
        qkv + (size_t)(b * SD + kb_lo + m) * NQKV + 1024 + kvh * HDD;          // tile0 key row
    unsigned long long vsrc =                                                   // async V src (bytes)
        (unsigned long long)(qkv + (size_t)(b * SD + kb_lo) * NQKV + 1280 + kvh * HDD)
        + (unsigned long long)((lane >> 3) * (NQKV * 2) + (lane & 7) * 16);
    const unsigned ldsv_dst = (unsigned)(unsigned long long)&ldsV[wv][0][0] + lane * 16;
    const unsigned short* lvrow = &ldsV[wv][hf * 16][0];                        // V frag read base
    const int dq0 = q0 + hf * 8 - m + WHALF;  // window test: 0 <= dq0 + r - kb <= 256

    for (int kb = kb_lo; kb < kb_hi; kb += 32) {
        // ---- kick off async V tile copy: 8 x 16B per lane = 4KB ----
#pragma unroll
        for (int c = 0; c < 8; ++c) {
            asm volatile("global_load_async_to_lds_b128 %0, %1, off"
                         :: "v"(ldsv_dst + c * 512),
                            "v"(vsrc + (unsigned long long)(c * 4 * NQKV * 2))
                         : "memory");
        }

        // ---- scores for two 16-key tiles (overlaps the async copy) ----
        v8f sc[2];
#pragma unroll
        for (int t = 0; t < 2; ++t) {
            const unsigned short* krow = krow0 + (size_t)t * 16 * NQKV;
            Frag kb0, kb1;  // B-layout: K-dim = hd, element e -> hd = chunk + (e>>3)*16 + hf*8 + (e&7)
            kb0.q[0] = *(const uint4*)(krow + hf * 8);
            kb0.q[1] = *(const uint4*)(krow + 16 + hf * 8);
            kb1.q[0] = *(const uint4*)(krow + 32 + hf * 8);
            kb1.q[1] = *(const uint4*)(krow + 48 + hf * 8);
            v8f z = v8f{0, 0, 0, 0, 0, 0, 0, 0};
            sc[t] = __builtin_amdgcn_wmma_f32_16x16x32_bf16(false, qa0.v, false, kb0.v, (short)0, z, false, false);
            sc[t] = __builtin_amdgcn_wmma_f32_16x16x32_bf16(false, qa1.v, false, kb1.v, (short)0, sc[t], false, false);
        }

        // prefetch next iteration's K rows (global_prefetch_b8)
        __builtin_prefetch(krow0 + (size_t)32 * NQKV, 0, 1);

        // ---- online softmax; alpha/P folded into the row loop ----
        const int d0 = dq0 - kb;              // valid0(r): (unsigned)(d0+r)    <= 256
        const bool in1 = (kb + 16 + m) < SD;  // valid1(r): (unsigned)(d0+r-16) <= 256 && in1
#pragma unroll
        for (int r = 0; r < 8; ++r) {
            bool v0 = (unsigned)(d0 + r) <= 2 * WHALF;
            bool v1 = ((unsigned)(d0 + r - 16) <= 2 * WHALF) && in1;
            float s0 = v0 ? sc[0][r] : -1e30f;   // scores already scaled via Q
            float s1 = v1 ? sc[1][r] : -1e30f;
            float mn = fmaxf(mi[r], rowmax16(fmaxf(s0, s1)));
            float al = __expf(mi[r] - mn);
            mi[r] = mn;
            float e0 = v0 ? __expf(s0 - mn) : 0.f;
            float e1 = v1 ? __expf(s1 - mn) : 0.f;
            li[r] = li[r] * al + rowsum16(e0 + e1);
            oacc[0][r] *= al; oacc[1][r] *= al; oacc[2][r] *= al; oacc[3][r] *= al;
            ldsP[wv][r + hf * 8][m]      = f2bf(e0);
            ldsP[wv][r + hf * 8][16 + m] = f2bf(e1);
        }
        asm volatile("s_wait_dscnt 0x0" ::: "memory");   // DS in-order per wave
        Frag pa;  // P in A-layout
        pa.q[0] = *(const uint4*)&ldsP[wv][m][hf * 16];
        pa.q[1] = *(const uint4*)&ldsP[wv][m][hf * 16 + 8];

        // ---- V tile ready? then 4 P*V WMMAs ----
        asm volatile("s_wait_asynccnt 0x0" ::: "memory");
#pragma unroll
        for (int j = 0; j < 4; ++j) {
            Frag vb;  // B-layout: K-dim = key, element e -> key = hf*16 + e; N col = j*16 + m
            const unsigned short* lv = lvrow + j * 16 + m;
#pragma unroll
            for (int e = 0; e < 16; ++e) vb.us[e] = lv[e * 64];
            oacc[j] = __builtin_amdgcn_wmma_f32_16x16x32_bf16(
                false, pa.v, false, vb.v, (short)0, oacc[j], false, false);
        }

        krow0 += (size_t)32 * NQKV;
        vsrc  += (unsigned long long)32 * NQKV * 2;
    }

    // ---- epilogue: normalize and store ctx (bf16, [b,s,hq,hd]) ----
#pragma unroll
    for (int r = 0; r < 8; ++r) {
        float rl = 1.f / li[r];
        unsigned short* crow =
            ctx + (size_t)(b * SD + q0 + r + hf * 8) * (HQD * HDD) + h * HDD + m;
#pragma unroll
        for (int j = 0; j < 4; ++j) crow[j * 16] = f2bf(oacc[j][r] * rl);
    }
}

// ---------------- host orchestration ----------------

extern "C" void kernel_launch(void* const* d_in, const int* in_sizes, int n_in,
                              void* d_out, int out_size, void* d_ws, size_t ws_size,
                              hipStream_t stream) {
    const float* x  = (const float*)d_in[0];
    const float* Wq = (const float*)d_in[1];
    const float* Wk = (const float*)d_in[2];
    const float* Wv = (const float*)d_in[3];
    const float* Wo = (const float*)d_in[4];

    // workspace layout (bf16/ushort elems): ~34.6 MB total
    unsigned short* ws    = (unsigned short*)d_ws;
    unsigned short* xb    = ws;                               // ROWS*D      = 4,194,304
    unsigned short* WqkvT = xb    + (size_t)ROWS * DD;        // 1536*1024   = 1,572,864
    unsigned short* WoT   = WqkvT + (size_t)NQKV * DD;        // 1024*1024   = 1,048,576
    unsigned short* qkv   = WoT   + (size_t)DD * DD;          // ROWS*1536   = 6,291,456
    unsigned short* ctx   = qkv   + (size_t)ROWS * NQKV;      // ROWS*1024   = 4,194,304

    const int nX = ROWS * DD;
    k_f32_to_bf16<<<(nX + 255) / 256, 256, 0, stream>>>(x, xb, nX);
    k_pack_wqkv<<<(NQKV * DD + 255) / 256, 256, 0, stream>>>(Wq, Wk, Wv, WqkvT);
    k_pack_wo<<<(DD * DD + 255) / 256, 256, 0, stream>>>(Wo, WoT);

    // fused QKV projection: 4096 x 1536 x 1024 -> (256 * 12) waves / 8 = 384 blocks
    k_gemm<1><<<384, 256, 0, stream>>>(xb, WqkvT, qkv, ROWS, NQKV, DD, DD, DD, NQKV);

    // RoPE over Q (16 heads, pre-scaled 0.125) + K (4 heads)
    k_rope<<<(ROWS * 20 * 32 + 255) / 256, 256, 0, stream>>>(qkv);

    // attention: B*HQ*(S/16) = 4096 waves -> 512 blocks
    k_attn<<<512, 256, 0, stream>>>(qkv, ctx);

    // output projection: 4096 x 1024 x 1024 -> (256 * 8) waves / 8 = 256 blocks, f32 out
    k_gemm<0><<<256, 256, 0, stream>>>(ctx, WoT, d_out, ROWS, DD, DD, DD, DD, DD);
}
